// AppearanceFineNet_52948356825731
// MI455X (gfx1250) — matrix operand order repaired and checked by reading, not compiled, not run on hardware
//
#include <hip/hip_runtime.h>

#define NE 200000
#define NN 10000
#define NFD 128
#define KCHUNK 128
#define MTILE 64
#define ASTR 136   // padded LDS row stride (ushorts) -> 272B, kills bank conflicts
#define BSTR 136

typedef __attribute__((ext_vector_type(16))) __bf16 v16bf;
typedef __attribute__((ext_vector_type(8)))  float  v8f;
typedef __attribute__((ext_vector_type(4)))  unsigned int v4u;
typedef __attribute__((ext_vector_type(4)))  int v4i;

// address-space qualified element types for the async-LDS builtin
typedef __attribute__((address_space(1))) v4i g_v4i;   // global
typedef __attribute__((address_space(3))) v4i l_v4i;   // LDS

union Frag16 { v16bf v; v4u q[2]; };

#if __has_builtin(__builtin_amdgcn_global_load_async_to_lds_b128)
#define HAVE_ASYNC_LDS 1
#else
#define HAVE_ASYNC_LDS 0
#endif

static __device__ __forceinline__ unsigned short f2bf(float f) {
    unsigned u = __float_as_uint(f);
    u += 0x7FFFu + ((u >> 16) & 1u);   // round-to-nearest-even
    return (unsigned short)(u >> 16);
}

static __device__ __forceinline__ float4 qmul4(float4 q, float4 r) {
    return make_float4(
        q.x * r.x - q.y * r.y - q.z * r.z - q.w * r.w,
        q.x * r.y + q.y * r.x + q.z * r.w - q.w * r.z,
        q.x * r.z - q.y * r.w + q.z * r.x + q.w * r.y,
        q.x * r.w + q.y * r.z - q.z * r.y + q.w * r.x);
}
static __device__ __forceinline__ float4 qinv(float4 q) {
    return make_float4(q.x, -q.y, -q.z, -q.w);
}

struct SegDesc {
    const float* base[6];
    int sel[6];     // 0 = gather by row, 1 = by col, 2 = by edge id
    int width[6];
    int nseg;
    int Ktot;
    int Kpad;       // multiple of 32, zero padded
};

// ---- W (K x 128, f32) -> Wt (128 x Kpad, bf16, transposed, zero padded) ----
__global__ void k_convert_w(const float* __restrict__ W,
                            unsigned short* __restrict__ Wt,
                            int K, int Kpad) {
    int i = blockIdx.x * blockDim.x + threadIdx.x;
    int total = NFD * Kpad;
    if (i >= total) return;
    int n = i / Kpad;
    int k = i - n * Kpad;
    Wt[n * Kpad + k] = (k < K) ? f2bf(W[k * NFD + n]) : (unsigned short)0;
}

// ---- eam = qmul(qmul(inv(x[col]), edge_attr), x[row]) ----
__global__ void k_eam(const float* __restrict__ xorg,
                      const float* __restrict__ eattr,
                      const int* __restrict__ row, const int* __restrict__ col,
                      float* __restrict__ eam) {
    int e = blockIdx.x * blockDim.x + threadIdx.x;
    if (e >= NE) return;
    float4 xc = reinterpret_cast<const float4*>(xorg)[col[e]];
    float4 xr = reinterpret_cast<const float4*>(xorg)[row[e]];
    float4 ea = reinterpret_cast<const float4*>(eattr)[e];
    reinterpret_cast<float4*>(eam)[e] = qmul4(qmul4(qinv(xc), ea), xr);
}

// ---- Edge GEMM: 64 edges x 128 outputs per block, bf16 WMMA 16x16x32.
//      B chunk staged in LDS once per block (async DMA on CDNA5), A gathered
//      into LDS; 8 WMMAs issued as a burst per K-step. ----
__global__ __launch_bounds__(128)
void k_edge_gemm(SegDesc sd, const unsigned short* __restrict__ Wt,
                 const float* __restrict__ bias,
                 const int* __restrict__ row, const int* __restrict__ col,
                 float* __restrict__ e_out, float* __restrict__ xsum,
                 float* __restrict__ cnt, int n_edges) {
    __shared__ __align__(16) unsigned short Alds[MTILE * ASTR];
    __shared__ __align__(16) unsigned short Blds[NFD * BSTR];

    const int tid   = threadIdx.x;
    const int wv    = tid >> 5;
    const int lane  = tid & 31;
    const int half  = (lane >> 4) & 1;
    const int l16   = lane & 15;
    const int ebase = blockIdx.x * MTILE;
    const int Kpad  = sd.Kpad;

    v8f acc[8] = {};

    // per-lane fragment base pointers (constant across K chunks)
    const unsigned short* a_base = &Alds[(wv * 16 + l16) * ASTR + half * 8];
    const unsigned short* b_base[8];
#pragma unroll
    for (int t = 0; t < 8; ++t)
        b_base[t] = &Blds[(t * 16 + l16) * BSTR + half * 8];

    for (int kc = 0; kc < Kpad; kc += KCHUNK) {
        int klen = Kpad - kc;
        if (klen > KCHUNK) klen = KCHUNK;

        __syncthreads();   // previous chunk fully consumed

        // ---- stage B chunk (128 x klen bf16) into LDS, async if available ----
        {
            const int nxf = klen >> 3;          // b128 transfers per row
            for (int id = tid; id < NFD * nxf; id += 128) {
                int n = id / nxf;
                int c = id - n * nxf;
                const unsigned short* src = Wt + (size_t)n * Kpad + kc + c * 8;
                unsigned short*       dst = &Blds[n * BSTR + c * 8];
#if HAVE_ASYNC_LDS
                __builtin_amdgcn_global_load_async_to_lds_b128(
                    (g_v4i*)src, (l_v4i*)dst, 0, 0);
#else
                *reinterpret_cast<v4u*>(dst) =
                    *reinterpret_cast<const v4u*>(src);
#endif
            }
        }

        // ---- cooperative gather of the concatenated A tile into LDS (bf16),
        //      overlaps with the in-flight async B copies ----
        for (int idx = tid; idx < MTILE * klen; idx += 128) {
            int m  = idx / klen;
            int kk = idx - m * klen;
            int kg = kc + kk;
            int e  = ebase + m;
            float v = 0.0f;
            if (kg < sd.Ktot && e < n_edges) {
                int s = 0, start = 0;
                while (s + 1 < sd.nseg && kg >= start + sd.width[s]) {
                    start += sd.width[s];
                    ++s;
                }
                int sel = sd.sel[s];
                int gi  = (sel == 0) ? row[e] : ((sel == 1) ? col[e] : e);
                v = sd.base[s][(long)gi * sd.width[s] + (kg - start)];
            }
            Alds[m * ASTR + kk] = f2bf(v);
        }

#if HAVE_ASYNC_LDS
#if __has_builtin(__builtin_amdgcn_s_wait_asynccnt)
        __builtin_amdgcn_s_wait_asynccnt(0);
#else
        asm volatile("s_wait_asynccnt 0x0" ::: "memory");
#endif
#endif
        __syncthreads();

        // ---- compute: per K-step load A frag + all 8 B frags, then burst ----
        for (int ks = 0; ks < klen; ks += 32) {
            Frag16 a;
            a.q[0] = *reinterpret_cast<const v4u*>(a_base + ks);
            a.q[1] = *reinterpret_cast<const v4u*>(a_base + ks + 16);
            Frag16 b[8];
#pragma unroll
            for (int t = 0; t < 8; ++t) {
                b[t].q[0] = *reinterpret_cast<const v4u*>(b_base[t] + ks);
                b[t].q[1] = *reinterpret_cast<const v4u*>(b_base[t] + ks + 16);
            }
#pragma unroll
            for (int t = 0; t < 8; ++t) {
                acc[t] = __builtin_amdgcn_wmma_f32_16x16x32_bf16(
                    false, a.v, false, b[t].v, (short)0, acc[t], false, false);
            }
        }
    }

    // epilogue: bias, write relu(m) to e_out, atomic segment-sum of pre-relu m
#pragma unroll
    for (int t = 0; t < 8; ++t) {
        int n = t * 16 + l16;
        float bn = bias[n];
#pragma unroll
        for (int r = 0; r < 8; ++r) {
            int m = wv * 16 + half * 8 + r;
            int e = ebase + m;
            if (e < n_edges) {
                float v = acc[t][r] + bn;
                e_out[(long)e * NFD + n] = v > 0.0f ? v : 0.0f;
                atomicAdd(&xsum[(long)row[e] * NFD + n], v);
            }
        }
    }
    if (l16 == 0) {
#pragma unroll
        for (int r = 0; r < 8; ++r) {
            int e = ebase + wv * 16 + half * 8 + r;
            if (e < n_edges) atomicAdd(&cnt[row[e]], 1.0f);
        }
    }
}

// ---- x = relu(xsum / max(cnt,1)) ----
__global__ void k_node_fin(const float* __restrict__ xsum,
                           const float* __restrict__ cnt,
                           float* __restrict__ x_out, int total) {
    int i = blockIdx.x * blockDim.x + threadIdx.x;
    if (i >= total) return;
    float c = cnt[i >> 7];
    c = c > 1.0f ? c : 1.0f;
    float v = xsum[i] / c;
    x_out[i] = v > 0.0f ? v : 0.0f;
}

// ---- x_final = l2norm(qmul(x4 @ lin1_w + lin1_b, x_org)) ----
__global__ void k_head(const float* __restrict__ x4,
                       const float* __restrict__ w, const float* __restrict__ b,
                       const float* __restrict__ xorg, float* __restrict__ x_out) {
    int i = blockIdx.x * blockDim.x + threadIdx.x;
    if (i >= NN) return;
    float p0 = b[0], p1 = b[1], p2 = b[2], p3 = b[3];
    const float* xr = x4 + (long)i * NFD;
    for (int k = 0; k < NFD; ++k) {
        float xv = xr[k];
        p0 += xv * w[k * 4 + 0];
        p1 += xv * w[k * 4 + 1];
        p2 += xv * w[k * 4 + 2];
        p3 += xv * w[k * 4 + 3];
    }
    float4 o = qmul4(make_float4(p0, p1, p2, p3),
                     reinterpret_cast<const float4*>(xorg)[i]);
    float nrm = sqrtf(o.x * o.x + o.y * o.y + o.z * o.z + o.w * o.w);
    nrm = fmaxf(nrm, 1e-12f);
    o.x /= nrm; o.y /= nrm; o.z /= nrm; o.w /= nrm;
    reinterpret_cast<float4*>(x_out)[i] = o;
}

// ---- out_res = e4 @ lin2_w + lin2_b ----
__global__ void k_res(const float* __restrict__ e4,
                      const float* __restrict__ w, const float* __restrict__ b,
                      float* __restrict__ out) {
    int e = blockIdx.x * blockDim.x + threadIdx.x;
    if (e >= NE) return;
    const float* r = e4 + (long)e * NFD;
    float s = b[0];
    for (int k = 0; k < NFD; ++k) s += r[k] * w[k];
    out[e] = s;
}

// ---- smooth-L1 loss of relative quaternions vs identity ----
__global__ void k_loss(const float* __restrict__ gt, const float* __restrict__ x,
                       const int* __restrict__ row, const int* __restrict__ col,
                       const float* __restrict__ beta, float* __restrict__ acc) {
    __shared__ float red[256];
    int e = blockIdx.x * blockDim.x + threadIdx.x;
    float s = 0.0f;
    if (e < NE) {
        float bt = beta[0];
        int ri = row[e], ci = col[e];
        float4 rg = qmul4(reinterpret_cast<const float4*>(gt)[ci],
                          qinv(reinterpret_cast<const float4*>(gt)[ri]));
        float4 rx = qmul4(reinterpret_cast<const float4*>(x)[ci],
                          qinv(reinterpret_cast<const float4*>(x)[ri]));
        float4 l1 = qmul4(qinv(rg), rx);
        float nrm = sqrtf(l1.x * l1.x + l1.y * l1.y + l1.z * l1.z + l1.w * l1.w);
        nrm = fmaxf(nrm, 1e-12f);
        float c[4] = { fabsf(l1.x / nrm - 1.0f), fabsf(l1.y / nrm),
                       fabsf(l1.z / nrm),        fabsf(l1.w / nrm) };
#pragma unroll
        for (int j = 0; j < 4; ++j)
            s += (c[j] < bt) ? 0.5f * c[j] * c[j] / bt : c[j] - 0.5f * bt;
    }
    red[threadIdx.x] = s;
    __syncthreads();
    for (int st = 128; st > 0; st >>= 1) {
        if ((int)threadIdx.x < st) red[threadIdx.x] += red[threadIdx.x + st];
        __syncthreads();
    }
    if (threadIdx.x == 0) atomicAdd(acc, red[0]);
}

__global__ void k_fin(const float* __restrict__ acc, const float* __restrict__ beta,
                      float* __restrict__ out_loss, float* __restrict__ out_beta) {
    if (blockIdx.x == 0 && threadIdx.x == 0) {
        out_loss[0] = acc[0] * (1.0f / (4.0f * (float)NE));
        out_beta[0] = beta[0];
    }
}

extern "C" void kernel_launch(void* const* d_in, const int* in_sizes, int n_in,
                              void* d_out, int out_size, void* d_ws, size_t ws_size,
                              hipStream_t stream) {
    const float* x_org     = (const float*)d_in[0];
    const float* edge_attr = (const float*)d_in[1];
    const float* gt_q      = (const float*)d_in[2];
    const float* beta      = (const float*)d_in[3];
    const float* node_feat = (const float*)d_in[4];
    const float* edge_feat = (const float*)d_in[5];
    const float* Ws[4] = { (const float*)d_in[6],  (const float*)d_in[8],
                           (const float*)d_in[10], (const float*)d_in[12] };
    const float* bs[4] = { (const float*)d_in[7],  (const float*)d_in[9],
                           (const float*)d_in[11], (const float*)d_in[13] };
    const float* lin1_w = (const float*)d_in[14];
    const float* lin1_b = (const float*)d_in[15];
    const float* lin2_w = (const float*)d_in[16];
    const float* lin2_b = (const float*)d_in[17];
    const int* eidx = (const int*)d_in[18];
    const int* row = eidx;
    const int* col = eidx + NE;

    float* out      = (float*)d_out;
    float* out_x    = out;                 // 10000*4
    float* out_loss = out + 40000;         // 1
    float* out_beta = out + 40001;         // 1
    float* out_res  = out + 40002;         // 200000
    float* out_xl[4];
    float* out_el[4];
    for (int i = 0; i < 4; ++i) {
        out_xl[i] = out + 240002 + (size_t)i * (NN * NFD);
        out_el[i] = out + 240002 + (size_t)4 * (NN * NFD) + (size_t)i * ((size_t)NE * NFD);
    }

    // workspace layout
    const int Kt[4] = { 396, 388, 768, 768 };
    const int Kp[4] = { 416, 416, 768, 768 };
    char* ws = (char*)d_ws;
    size_t o = 0;
    unsigned short* Wt[4];
    for (int i = 0; i < 4; ++i) { Wt[i] = (unsigned short*)(ws + o); o += (size_t)NFD * Kp[i] * 2; }
    o = (o + 255) & ~(size_t)255;
    float* eam  = (float*)(ws + o); o += (size_t)NE * 4 * 4;
    float* xsum = (float*)(ws + o); o += (size_t)NN * NFD * 4;
    float* cnt  = (float*)(ws + o); o += (size_t)NN * 4;
    float* lacc = (float*)(ws + o); o += 4;

    // weight conversion (f32 KxN -> bf16 NxKpad transposed)
    for (int l = 0; l < 4; ++l) {
        int tot = NFD * Kp[l];
        k_convert_w<<<(tot + 255) / 256, 256, 0, stream>>>(Ws[l], Wt[l], Kt[l], Kp[l]);
    }
    k_eam<<<(NE + 255) / 256, 256, 0, stream>>>(x_org, edge_attr, row, col, eam);

    // segment descriptors per layer
    SegDesc sd[4];
    for (int l = 0; l < 4; ++l)
        for (int j = 0; j < 6; ++j) { sd[l].base[j] = nullptr; sd[l].sel[j] = 2; sd[l].width[j] = 0; }

    // L1: [node_feat,x_org][row] | [node_feat,x_org][col] | [edge_feat, eam]
    sd[0].nseg = 6; sd[0].Ktot = Kt[0]; sd[0].Kpad = Kp[0];
    sd[0].base[0] = node_feat; sd[0].sel[0] = 0; sd[0].width[0] = 128;
    sd[0].base[1] = x_org;     sd[0].sel[1] = 0; sd[0].width[1] = 4;
    sd[0].base[2] = node_feat; sd[0].sel[2] = 1; sd[0].width[2] = 128;
    sd[0].base[3] = x_org;     sd[0].sel[3] = 1; sd[0].width[3] = 4;
    sd[0].base[4] = edge_feat; sd[0].sel[4] = 2; sd[0].width[4] = 128;
    sd[0].base[5] = eam;       sd[0].sel[5] = 2; sd[0].width[5] = 4;
    // L2: x1[row] | x1[col] | [eam, e1]
    sd[1].nseg = 4; sd[1].Ktot = Kt[1]; sd[1].Kpad = Kp[1];
    sd[1].base[0] = out_xl[0]; sd[1].sel[0] = 0; sd[1].width[0] = 128;
    sd[1].base[1] = out_xl[0]; sd[1].sel[1] = 1; sd[1].width[1] = 128;
    sd[1].base[2] = eam;       sd[1].sel[2] = 2; sd[1].width[2] = 4;
    sd[1].base[3] = out_el[0]; sd[1].sel[3] = 2; sd[1].width[3] = 128;
    // L3: [x2,x1][row] | [x2,x1][col] | [e2,e1]
    sd[2].nseg = 6; sd[2].Ktot = Kt[2]; sd[2].Kpad = Kp[2];
    sd[2].base[0] = out_xl[1]; sd[2].sel[0] = 0; sd[2].width[0] = 128;
    sd[2].base[1] = out_xl[0]; sd[2].sel[1] = 0; sd[2].width[1] = 128;
    sd[2].base[2] = out_xl[1]; sd[2].sel[2] = 1; sd[2].width[2] = 128;
    sd[2].base[3] = out_xl[0]; sd[2].sel[3] = 1; sd[2].width[3] = 128;
    sd[2].base[4] = out_el[1]; sd[2].sel[4] = 2; sd[2].width[4] = 128;
    sd[2].base[5] = out_el[0]; sd[2].sel[5] = 2; sd[2].width[5] = 128;
    // L4: [x3,x2][row] | [x3,x2][col] | [e3,e2]
    sd[3].nseg = 6; sd[3].Ktot = Kt[3]; sd[3].Kpad = Kp[3];
    sd[3].base[0] = out_xl[2]; sd[3].sel[0] = 0; sd[3].width[0] = 128;
    sd[3].base[1] = out_xl[1]; sd[3].sel[1] = 0; sd[3].width[1] = 128;
    sd[3].base[2] = out_xl[2]; sd[3].sel[2] = 1; sd[3].width[2] = 128;
    sd[3].base[3] = out_xl[1]; sd[3].sel[3] = 1; sd[3].width[3] = 128;
    sd[3].base[4] = out_el[2]; sd[3].sel[4] = 2; sd[3].width[4] = 128;
    sd[3].base[5] = out_el[1]; sd[3].sel[5] = 2; sd[3].width[5] = 128;

    for (int l = 0; l < 4; ++l) {
        (void)hipMemsetAsync(xsum, 0, (size_t)NN * NFD * 4, stream);
        (void)hipMemsetAsync(cnt,  0, (size_t)NN * 4, stream);
        k_edge_gemm<<<(NE + MTILE - 1) / MTILE, 128, 0, stream>>>(
            sd[l], Wt[l], bs[l], row, col, out_el[l], xsum, cnt, NE);
        k_node_fin<<<(NN * NFD + 255) / 256, 256, 0, stream>>>(xsum, cnt, out_xl[l], NN * NFD);
    }

    k_head<<<(NN + 255) / 256, 256, 0, stream>>>(out_xl[3], lin1_w, lin1_b, x_org, out_x);
    k_res<<<(NE + 255) / 256, 256, 0, stream>>>(out_el[3], lin2_w, lin2_b, out_res);
    (void)hipMemsetAsync(lacc, 0, 4, stream);
    k_loss<<<(NE + 255) / 256, 256, 0, stream>>>(gt_q, out_x, row, col, beta, lacc);
    k_fin<<<1, 1, 0, stream>>>(lacc, beta, out_loss, out_beta);
}
